// LieNet_16535624090422
// MI455X (gfx1250) — matrix operand order repaired and checked by reading, not compiled, not run on hardware
//
#include <hip/hip_runtime.h>
#include <hip/hip_bf16.h>
#include <math.h>

#define S_DIM 512
#define B_DIM 256
#define D_DIM 256

typedef __attribute__((ext_vector_type(16))) __bf16 v16bf;
typedef __attribute__((ext_vector_type(8)))  __bf16 v8bf;
typedef __attribute__((ext_vector_type(8)))  float  v8f;

__device__ __forceinline__ float gelu_exact(float x) {
    // exact gelu: 0.5*x*(1+erf(x/sqrt(2)))
    return 0.5f * x * (1.0f + erff(x * 0.70710678118654752440f));
}

__global__ void f32_to_bf16_kernel(const float* __restrict__ in,
                                   __bf16* __restrict__ out, int n) {
    int i = blockIdx.x * blockDim.x + threadIdx.x;
    if (i < n) out[i] = (__bf16)in[i];
}

// h = gelu(srcb @ map_w^T + map_b), srcb bf16 [S*B, D], W bf16 [D, D] row-major.
// Block = 128 threads (4 waves). Each wave: 32(M) x 64(N) tile (2x4 WMMA tiles).
// grid = (M/128, N/64).
__global__ void __launch_bounds__(128) map_gemm_kernel(
    const __bf16* __restrict__ srcb, const __bf16* __restrict__ Wb,
    const float* __restrict__ bias, __bf16* __restrict__ hout)
{
    const int lane = threadIdx.x & 31;
    const int wave = threadIdx.x >> 5;
    const int half = lane >> 4;
    const int lm   = lane & 15;
    const int rowbase = blockIdx.x * 128 + wave * 32;
    const int colbase = blockIdx.y * 64;

    // Pointers with the per-lane layout offsets baked in; all loop loads use
    // compile-time immediate offsets only.
    const __bf16* ar[2];
    #pragma unroll
    for (int mi = 0; mi < 2; ++mi)
        ar[mi] = srcb + (size_t)(rowbase + mi * 16 + lm) * D_DIM + 16 * half;
    const __bf16* wr[4];
    #pragma unroll
    for (int ni = 0; ni < 4; ++ni)
        wr[ni] = Wb + (size_t)(colbase + ni * 16 + lm) * D_DIM + 16 * half;

    v8f acc[2][4] = {};

    #pragma unroll
    for (int kb = 0; kb < 8; ++kb) {
        v16bf a[2];
        #pragma unroll
        for (int mi = 0; mi < 2; ++mi) {
            // A 16x32 bf16 layout: half=0 -> K {0..15}; half=1 -> {16..31}
            v8bf lo = *(const v8bf*)(ar[mi] + kb * 32);
            v8bf hi = *(const v8bf*)(ar[mi] + kb * 32 + 8);
            a[mi] = __builtin_shufflevector(lo, hi,
                     0,1,2,3,4,5,6,7,8,9,10,11,12,13,14,15);
        }
        v16bf b[4];
        #pragma unroll
        for (int ni = 0; ni < 4; ++ni)
            b[ni] = *(const v16bf*)(wr[ni] + kb * 32); // B 32x16: contiguous 16 K per lane
        #pragma unroll
        for (int ni = 0; ni < 4; ++ni)
            #pragma unroll
            for (int mi = 0; mi < 2; ++mi)
                acc[mi][ni] = __builtin_amdgcn_wmma_f32_16x16x32_bf16(
                    false, a[mi], false, b[ni], (short)0, acc[mi][ni], false, false);
    }

    #pragma unroll
    for (int mi = 0; mi < 2; ++mi)
        #pragma unroll
        for (int ni = 0; ni < 4; ++ni) {
            const int n = colbase + ni * 16 + lm;
            const float bn = bias[n];
            #pragma unroll
            for (int r = 0; r < 8; ++r) {
                const int m = rowbase + mi * 16 + r + 8 * half;
                float g = gelu_exact(acc[mi][ni][r] + bn);
                hout[(size_t)m * D_DIM + n] = (__bf16)g;
            }
        }
}

// Generic blacket: out = gelu(A[t+sA] @ W[:, :D]^T + B[t+sB] @ W[:, D:]^T + bias)
// Out-of-range s-index => zero operand. Optional: +addsrc[t+sAdd] (after gelu),
// f32 store/accumulate, bf16 store. W bf16 [D, 2D] row-major.
// Block = 128 threads (4 waves); each wave 32x64; grid = (2, 4, S - t0).
__global__ void __launch_bounds__(128) blacket_kernel(
    const __bf16* __restrict__ Abuf, int sA,
    const __bf16* __restrict__ Bbuf, int sB,
    const __bf16* __restrict__ Wb,
    const float* __restrict__ bias,
    const __bf16* __restrict__ addsrc, int sAdd,
    float* __restrict__ outf, int accumulate,
    __bf16* __restrict__ outb, int t0)
{
    const int lane = threadIdx.x & 31;
    const int wave = threadIdx.x >> 5;
    const int half = lane >> 4;
    const int lm   = lane & 15;
    const int t    = t0 + blockIdx.z;
    const int rowbase = blockIdx.x * 128 + wave * 32;   // row within the t block (0..255)
    const int colbase = blockIdx.y * 64;

    const int  sa = t + sA;
    const int  sb = t + sB;
    const bool va = (sa >= 0) && (sa < S_DIM);
    const bool vb = (sb >= 0) && (sb < S_DIM);

    const __bf16* ar[2];
    const __bf16* br[2];
    #pragma unroll
    for (int mi = 0; mi < 2; ++mi) {
        const size_t r = (size_t)(rowbase + mi * 16 + lm) * D_DIM + 16 * half;
        ar[mi] = Abuf + (size_t)sa * (B_DIM * D_DIM) + r;
        br[mi] = Bbuf + (size_t)sb * (B_DIM * D_DIM) + r;
    }
    const __bf16* wr[4];
    #pragma unroll
    for (int ni = 0; ni < 4; ++ni)
        wr[ni] = Wb + (size_t)(colbase + ni * 16 + lm) * (2 * D_DIM) + 16 * half;

    v8f acc[2][4] = {};
    // K = 512: kb 0..7 from A-source (weight cols 0..255), kb 8..15 from B-source.
    #pragma unroll
    for (int kb = 0; kb < 16; ++kb) {
        const bool second = (kb >= 8);
        const int  klocal = (second ? (kb - 8) : kb) * 32;
        const bool valid  = second ? vb : va;

        v16bf a[2];
        #pragma unroll
        for (int mi = 0; mi < 2; ++mi) {
            a[mi] = (v16bf){};
            if (valid) {
                const __bf16* p = (second ? br[mi] : ar[mi]) + klocal;
                v8bf lo = *(const v8bf*)(p);
                v8bf hi = *(const v8bf*)(p + 8);
                a[mi] = __builtin_shufflevector(lo, hi,
                         0,1,2,3,4,5,6,7,8,9,10,11,12,13,14,15);
            }
        }
        v16bf b[4];
        #pragma unroll
        for (int ni = 0; ni < 4; ++ni)
            b[ni] = *(const v16bf*)(wr[ni] + kb * 32);
        #pragma unroll
        for (int ni = 0; ni < 4; ++ni)
            #pragma unroll
            for (int mi = 0; mi < 2; ++mi)
                acc[mi][ni] = __builtin_amdgcn_wmma_f32_16x16x32_bf16(
                    false, a[mi], false, b[ni], (short)0, acc[mi][ni], false, false);
    }

    const int  sadd = t + sAdd;
    const bool vadd = (addsrc != nullptr) && (sadd >= 0) && (sadd < S_DIM);
    #pragma unroll
    for (int mi = 0; mi < 2; ++mi)
        #pragma unroll
        for (int ni = 0; ni < 4; ++ni) {
            const int n = colbase + ni * 16 + lm;
            const float bn = bias[n];
            #pragma unroll
            for (int r = 0; r < 8; ++r) {
                const int m = rowbase + mi * 16 + r + 8 * half;
                const size_t oidx = ((size_t)t * B_DIM + m) * D_DIM + n;
                float g = gelu_exact(acc[mi][ni][r] + bn);
                if (outb) outb[oidx] = (__bf16)g;
                if (outf) {
                    if (accumulate) {
                        outf[oidx] += g;
                    } else {
                        float v = g;
                        if (vadd) v += (float)addsrc[((size_t)sadd * B_DIM + m) * D_DIM + n];
                        outf[oidx] = v;
                    }
                }
            }
        }
}

// In-place cumsum along s: one thread per (b,d) column, coalesced across d.
__global__ void cumsum_kernel(float* __restrict__ out) {
    const int c = blockIdx.x * blockDim.x + threadIdx.x; // 0 .. B*D-1
    float acc = 0.0f;
    for (int s = 0; s < S_DIM; ++s) {
        const size_t i = (size_t)s * (B_DIM * D_DIM) + c;
        acc += out[i];
        out[i] = acc;
    }
}

extern "C" void kernel_launch(void* const* d_in, const int* in_sizes, int n_in,
                              void* d_out, int out_size, void* d_ws, size_t ws_size,
                              hipStream_t stream)
{
    (void)in_sizes; (void)n_in; (void)out_size; (void)ws_size;
    const float* src   = (const float*)d_in[0];
    const float* map_w = (const float*)d_in[1];
    const float* map_b = (const float*)d_in[2];
    const float* bl_w  = (const float*)d_in[3];
    const float* bl_b  = (const float*)d_in[4];
    float* out = (float*)d_out;

    const size_t NE = (size_t)S_DIM * B_DIM * D_DIM; // 33,554,432 elements
    char* ws = (char*)d_ws;
    __bf16* h    = (__bf16*)ws; ws += NE * sizeof(__bf16);
    __bf16* P    = (__bf16*)ws; ws += NE * sizeof(__bf16);
    __bf16* Q    = (__bf16*)ws; ws += NE * sizeof(__bf16);
    __bf16* srcb = (__bf16*)ws; ws += NE * sizeof(__bf16);
    __bf16* mw   = (__bf16*)ws; ws += (size_t)D_DIM * D_DIM * sizeof(__bf16);
    __bf16* bw   = (__bf16*)ws; ws += (size_t)D_DIM * 2 * D_DIM * sizeof(__bf16);

    // bf16 conversions (weights + src) so GEMM hot loops are pure load+WMMA
    f32_to_bf16_kernel<<<(D_DIM * D_DIM + 255) / 256, 256, 0, stream>>>(map_w, mw, D_DIM * D_DIM);
    f32_to_bf16_kernel<<<(2 * D_DIM * D_DIM + 255) / 256, 256, 0, stream>>>(bl_w, bw, 2 * D_DIM * D_DIM);
    f32_to_bf16_kernel<<<(int)((NE + 255) / 256), 256, 0, stream>>>(src, srcb, (int)NE);

    // h = gelu(src @ map_w^T + map_b): grid (131072/128, 256/64)
    map_gemm_kernel<<<dim3((S_DIM * B_DIM) / 128, D_DIM / 64), 128, 0, stream>>>(srcb, mw, map_b, h);

    // P[t] = bl(h[t-1], h[t]);  out[t] = x[t] + P[t]  (x[t] = h[t-1], zero at t=0)
    blacket_kernel<<<dim3(2, 4, S_DIM), 128, 0, stream>>>(
        h, -1, h, 0, bw, bl_b, h, -1, out, 0, P, 0);

    // Q[t] = bl(h[t], h[t-2])  for t >= 2
    blacket_kernel<<<dim3(2, 4, S_DIM - 2), 128, 0, stream>>>(
        h, 0, h, -2, bw, bl_b, nullptr, 0, nullptr, 0, Q, 2);

    // out[t] += bl(h[t-2], P[t])      == bl(a, bl(b,c))
    blacket_kernel<<<dim3(2, 4, S_DIM - 2), 128, 0, stream>>>(
        h, -2, P, 0, bw, bl_b, nullptr, 0, out, 1, nullptr, 2);

    // out[t] += bl(h[t-1], Q[t])      == bl(b, bl(c,a))
    blacket_kernel<<<dim3(2, 4, S_DIM - 2), 128, 0, stream>>>(
        h, -1, Q, 0, bw, bl_b, nullptr, 0, out, 1, nullptr, 2);

    // out[t] += bl(h[t], P[t-1])      == bl(c, bl(a,b))
    blacket_kernel<<<dim3(2, 4, S_DIM - 2), 128, 0, stream>>>(
        h, 0, P, -1, bw, bl_b, nullptr, 0, out, 1, nullptr, 2);

    // cumsum along s, in place
    cumsum_kernel<<<dim3((B_DIM * D_DIM) / 256), 256, 0, stream>>>(out);
}
